// GCNModel_57440892617189
// MI455X (gfx1250) — matrix operand order repaired and checked by reading, not compiled, not run on hardware
//
#include <hip/hip_runtime.h>
#include <hip/hip_bf16.h>
#include <math.h>

typedef __attribute__((ext_vector_type(16))) _Float16 v16h;
typedef __attribute__((ext_vector_type(8)))  _Float16 v8h;
typedef __attribute__((ext_vector_type(4)))  _Float16 v4h;
typedef __attribute__((ext_vector_type(8)))  float    v8f;
typedef __attribute__((ext_vector_type(4)))  float    v4f;

#define HID 128
#define DSW 132   // padded f32 LDS row stride (col%64 and (col+32)%64 bank sets -> conflict-free)

// ---------- helpers ----------
static __device__ __forceinline__ unsigned encF(float x) {
  unsigned b = __float_as_uint(x);
  return (b & 0x80000000u) ? ~b : (b | 0x80000000u);   // order-preserving encoding
}
static __device__ __forceinline__ float decF(unsigned u) {
  return (u & 0x80000000u) ? __uint_as_float(u ^ 0x80000000u) : __uint_as_float(~u);
}
static __device__ __forceinline__ float eluf(float x) { return x > 0.f ? x : expm1f(x); }

// ---------- elementwise / setup kernels ----------
__global__ void k_zero_f32(float* __restrict__ p, long n) {
  long i = (long)blockIdx.x * blockDim.x + threadIdx.x;
  if (i < n) p[i] = 0.f;
}

__global__ void k_degrees(const int* __restrict__ src, const int* __restrict__ dst,
                          float* __restrict__ degOut, float* __restrict__ degIn, int E) {
  int e = blockIdx.x * blockDim.x + threadIdx.x;
  if (e < E) {
    atomicAdd(&degOut[src[e]], 1.f);
    atomicAdd(&degIn[dst[e]], 1.f);
  }
}

__global__ void k_norms(const float* __restrict__ degOut, const float* __restrict__ degIn,
                        float* __restrict__ outN, float* __restrict__ inN, int N) {
  int n = blockIdx.x * blockDim.x + threadIdx.x;
  if (n < N) {
    outN[n] = rsqrtf(fmaxf(degOut[n], 1.f));
    inN[n]  = rsqrtf(fmaxf(degIn[n], 1.f));
  }
}

// layer-1 dense transform: h[n][c] = (f0*s)*W1[0][c] + (f1*s)*W1[1][c]
__global__ void k_layer1(const float* __restrict__ feats, const float* __restrict__ outN,
                         const float* __restrict__ W1, float* __restrict__ h, long total) {
  long i = (long)blockIdx.x * blockDim.x + threadIdx.x;
  if (i >= total) return;
  int n = (int)(i >> 7), c = (int)(i & 127);
  float s  = outN[n];
  float f0 = feats[2 * n] * s, f1 = feats[2 * n + 1] * s;
  h[i] = f0 * W1[c] + f1 * W1[HID + c];
}

// edge-parallel SpMM: 32 lanes per edge, float4 gather + scatter atomics
__global__ void k_spmm(const int* __restrict__ src, const int* __restrict__ dst,
                       const float* __restrict__ h, float* __restrict__ agg, int E) {
  int tid = blockIdx.x * blockDim.x + threadIdx.x;
  int e = tid >> 5;
  if (e >= E) return;
  int c4 = (tid & 31) << 2;
  int s = src[e], d = dst[e];
  v4f v = *(const v4f*)(h + (size_t)s * HID + c4);
  float* base = agg + (size_t)d * HID + c4;
  atomicAdd(base + 0, v[0]);
  atomicAdd(base + 1, v[1]);
  atomicAdd(base + 2, v[2]);
  atomicAdd(base + 3, v[3]);
}

// agg * in_norm + bias -> tanh  (in place)
__global__ void k_post(float* __restrict__ buf, const float* __restrict__ inN,
                       const float* __restrict__ bias, long total) {
  long i = (long)blockIdx.x * blockDim.x + threadIdx.x;
  if (i >= total) return;
  int n = (int)(i >> 7), c = (int)(i & 127);
  buf[i] = tanhf(buf[i] * inN[n] + bias[c]);
}

// W2[128x128] fp32 -> transposed f16: Wt[n][k] = W2[k][n]
__global__ void k_w2f16t(const float* __restrict__ W, _Float16* __restrict__ Wt) {
  int i = blockIdx.x * blockDim.x + threadIdx.x;    // 16384
  int n = i >> 7, k = i & 127;
  Wt[n * HID + k] = (_Float16)W[k * HID + n];
}

// ---------- WMMA GEMM: D[N x 128] = (X * rowScale) @ W  (W transposed f16) ----------
// Also zeroes X after staging (X becomes the next SpMM accumulator).
__global__ __launch_bounds__(256) void k_gemm_wmma(
    float* X,                                // read, then zeroed (fused agg-clear)
    const float* __restrict__ rowScale,
    const _Float16* __restrict__ Wt, float* __restrict__ D, int nRows)
{
  __shared__ __align__(16) unsigned char smem[16 * DSW * sizeof(float)]; // 8.25 KB
  _Float16* As = (_Float16*)smem;            // f16 A panel, stride HID
  float*    Ds = (float*)smem;               // f32 C panel, stride DSW

  const int mBase = blockIdx.x * 16;
  const int t = threadIdx.x;

  // ---- stage A: 16 rows x 128 cols, scale by out_norm, cvt f16; zero X behind us
  for (int i = t; i < 16 * 32; i += 256) {          // 512 float4 chunks
    int r = i >> 5;
    int k4 = (i & 31) << 2;
    int row = mBase + r;
    v4f v = {0.f, 0.f, 0.f, 0.f};
    if (row < nRows) {
      float* xp = X + (size_t)row * HID + k4;
      v = *(const v4f*)xp;
      float s = rowScale[row];
      v[0] *= s; v[1] *= s; v[2] *= s; v[3] *= s;
      *(v4f*)xp = (v4f){0.f, 0.f, 0.f, 0.f};        // fused zero for next SpMM pass
    }
    v4h hv;
    hv[0] = (_Float16)v[0]; hv[1] = (_Float16)v[1];
    hv[2] = (_Float16)v[2]; hv[3] = (_Float16)v[3];
    *(v4h*)(&As[r * HID + k4]) = hv;
  }
  __syncthreads();

  const int wave  = t >> 5;       // 8 waves -> 8 N-tiles of 16
  const int lane  = t & 31;
  const int r     = lane & 15;
  const int hi    = lane >> 4;
  const int nBase = wave * 16;

  v8f acc = {};
  const _Float16* Arow = &As[r * HID];
  const _Float16* Bcol = Wt + (size_t)(nBase + r) * HID;

#pragma unroll
  for (int kb = 0; kb < HID; kb += 32) {
    // A 16x32 fragment: lane holds row r; K = {kb+hi*8..+7} and {kb+16+hi*8..+7}
    v8h a0 = *(const v8h*)(Arow + kb + hi * 8);
    v8h a1 = *(const v8h*)(Arow + kb + 16 + hi * 8);
    v16h a = __builtin_shufflevector(a0, a1, 0,1,2,3,4,5,6,7,8,9,10,11,12,13,14,15);
    // B 32x16 fragment: lane holds col nBase+r; K = {kb+hi*8..+7} and {kb+16+hi*8..+7}
    v8h b0 = *(const v8h*)(Bcol + kb + hi * 8);
    v8h b1 = *(const v8h*)(Bcol + kb + 16 + hi * 8);
    v16h b = __builtin_shufflevector(b0, b1, 0,1,2,3,4,5,6,7,8,9,10,11,12,13,14,15);
    acc = __builtin_amdgcn_wmma_f32_16x16x32_f16(false, a, false, b,
                                                 (short)0, acc, false, false);
  }
  __syncthreads();   // A panel reads done; smem is reused as f32 C panel

  // C/D fragment layout: VGPR v -> (M = hi*8 + v, Ncol = nBase + r)
#pragma unroll
  for (int v = 0; v < 8; ++v)
    Ds[(hi * 8 + v) * DSW + nBase + r] = acc[v];
  __syncthreads();

  // ---- coalesced b128 epilogue
  for (int i = t; i < 16 * 32; i += 256) {
    int r2 = i >> 5;
    int k4 = (i & 31) << 2;
    int row = mBase + r2;
    if (row < nRows)
      *(v4f*)(D + (size_t)row * HID + k4) = *(const v4f*)(&Ds[r2 * DSW + k4]);
  }
}

// ---------- pooling ----------
__global__ void k_pool_init(unsigned* __restrict__ pool, int total) {
  int i = blockIdx.x * blockDim.x + threadIdx.x;
  if (i < total) pool[i] = encF(-3.0e38f);
}

__global__ void k_pool_max(const float* __restrict__ h, const int* __restrict__ gid,
                           unsigned* __restrict__ pool, long total) {
  long i = (long)blockIdx.x * blockDim.x + threadIdx.x;
  if (i >= total) return;
  int n = (int)(i >> 7), c = (int)(i & 127);
  atomicMax(&pool[(size_t)gid[n] * HID + c], encF(h[i]));
}

// ---------- fused classifier: one block per graph ----------
__global__ __launch_bounds__(128) void k_classifier(
    const unsigned* __restrict__ pool,
    const float* __restrict__ Wc1, const float* __restrict__ bc1,
    const float* __restrict__ Wc2, const float* __restrict__ bc2,
    const float* __restrict__ Wc3, const float* __restrict__ bc3,
    float* __restrict__ out)
{
  __shared__ float xr[HID];
  __shared__ float h1s[HID];
  __shared__ float h2s[32];
  int g = blockIdx.x, t = threadIdx.x;

  xr[t] = decF(pool[(size_t)g * HID + t]);
  __syncthreads();

  float acc = bc1[t];
#pragma unroll 8
  for (int k = 0; k < HID; ++k) acc += xr[k] * Wc1[k * HID + t];
  h1s[t] = eluf(acc);
  __syncthreads();

  if (t < 32) {
    float a2 = bc2[t];
#pragma unroll 8
    for (int k = 0; k < HID; ++k) a2 += h1s[k] * Wc2[k * 32 + t];
    h2s[t] = eluf(a2);
  }
  __syncthreads();

  if (t == 0) {
    float a3 = bc3[0];
#pragma unroll
    for (int k = 0; k < 32; ++k) a3 += h2s[k] * Wc3[k];
    out[g] = a3;
  }
}

// ---------- host launch ----------
extern "C" void kernel_launch(void* const* d_in, const int* in_sizes, int n_in,
                              void* d_out, int out_size, void* d_ws, size_t ws_size,
                              hipStream_t stream) {
  const float* feats = (const float*)d_in[0];
  const int*   src   = (const int*)d_in[1];
  const int*   dst   = (const int*)d_in[2];
  const int*   gids  = (const int*)d_in[3];
  const float* W1    = (const float*)d_in[4];
  const float* b1    = (const float*)d_in[5];
  const float* W2    = (const float*)d_in[6];
  const float* b2    = (const float*)d_in[7];
  const float* Wc1   = (const float*)d_in[8];
  const float* bc1   = (const float*)d_in[9];
  const float* Wc2   = (const float*)d_in[10];
  const float* bc2   = (const float*)d_in[11];
  const float* Wc3   = (const float*)d_in[12];
  const float* bc3   = (const float*)d_in[13];

  const int  N = in_sizes[0] / 2;
  const int  E = in_sizes[1];
  const int  G = out_size;                   // 128 graphs, 1 scalar each
  const long NH = (long)N * HID;

  // carve workspace
  char* ws = (char*)d_ws;
  size_t off = 0;
  auto carve = [&](size_t bytes) -> void* {
    void* p = ws + off;
    off = (off + bytes + 255) & ~(size_t)255;
    return p;
  };
  float*     deg    = (float*)carve((size_t)2 * N * sizeof(float));  // [degOut | degIn]
  float*     degOut = deg;
  float*     degIn  = deg + N;
  float*     outN   = (float*)carve((size_t)N * sizeof(float));
  float*     inN    = (float*)carve((size_t)N * sizeof(float));
  float*     bufA   = (float*)carve(NH * sizeof(float));
  float*     bufB   = (float*)carve(NH * sizeof(float));
  _Float16*  Wt     = (_Float16*)carve((size_t)HID * HID * sizeof(_Float16));
  unsigned*  pool   = (unsigned*)carve((size_t)G * HID * sizeof(unsigned));
  (void)ws_size; (void)n_in;

  const int T = 256;
  dim3 blk(T);
  auto blocksFor = [&](long n) { return dim3((unsigned)((n + T - 1) / T)); };

  // degrees + norms
  k_zero_f32<<<blocksFor((long)2 * N), blk, 0, stream>>>(deg, (long)2 * N);
  k_degrees<<<blocksFor(E), blk, 0, stream>>>(src, dst, degOut, degIn, E);
  k_norms<<<blocksFor(N), blk, 0, stream>>>(degOut, degIn, outN, inN, N);

  // ---- layer 1 ----
  k_layer1<<<blocksFor(NH), blk, 0, stream>>>(feats, outN, W1, bufA, NH);
  k_zero_f32<<<blocksFor(NH), blk, 0, stream>>>(bufB, NH);
  k_spmm<<<blocksFor((long)E * 32), blk, 0, stream>>>(src, dst, bufA, bufB, E);
  k_post<<<blocksFor(NH), blk, 0, stream>>>(bufB, inN, b1, NH);   // bufB = h1

  // ---- layer 2 (GEMM zeroes bufB behind itself -> ready as SpMM accumulator) ----
  k_w2f16t<<<blocksFor(HID * HID), blk, 0, stream>>>(W2, Wt);
  k_gemm_wmma<<<dim3((N + 15) / 16), blk, 0, stream>>>(bufB, outN, Wt, bufA, N);
  k_spmm<<<blocksFor((long)E * 32), blk, 0, stream>>>(src, dst, bufA, bufB, E);
  k_post<<<blocksFor(NH), blk, 0, stream>>>(bufB, inN, b2, NH);   // bufB = h2

  // ---- max pooling ----
  k_pool_init<<<blocksFor(G * HID), blk, 0, stream>>>(pool, G * HID);
  k_pool_max<<<blocksFor(NH), blk, 0, stream>>>(bufB, gids, pool, NH);

  // ---- classifier ----
  k_classifier<<<dim3(G), dim3(HID), 0, stream>>>(pool, Wc1, bc1, Wc2, bc2, Wc3, bc3,
                                                  (float*)d_out);
}
